// AttnDecoder_25426206392360
// MI455X (gfx1250) — matrix operand order repaired and checked by reading, not compiled, not run on hardware
//
#include <hip/hip_runtime.h>
#include <hip/hip_bf16.h>
#include <math.h>

// ---------------------------------------------------------------------------
// AttnDecoder for MI455X (gfx1250, wave32).
// Roofline: dominated by W_o GEMV (262MB fp32/step x 32 steps = 8.4GB HBM).
// Fix: one-time fp32->bf16 conversion of W_o/W_ih/W_hh into workspace; bf16
// W_o = 131MB < 192MB L2 => steady-state L2 resident. GEMVs are executed as
// wave-level v_wmma_f32_16x16x32_bf16 with the x-vector replicated over the
// 16 A-rows (memory-bound, so 1/16 MACs utilization is free).
// ---------------------------------------------------------------------------

typedef __attribute__((ext_vector_type(16))) __bf16  v16bf;
typedef __attribute__((ext_vector_type(8)))  float   v8f;
typedef __attribute__((ext_vector_type(4)))  unsigned int u32x4;

union BFrag { u32x4 q[2]; v16bf v; };

#define V_SIZE 32000
#define E_SIZE 512
#define H_SIZE 1024
#define S_SIZE 64
#define T_STEPS 32
#define SOS_TOK 1

__device__ __forceinline__ unsigned short f2bf(float f) {
    unsigned u = __float_as_uint(f);
    unsigned r = u + 0x7FFFu + ((u >> 16) & 1u);   // round-to-nearest-even
    return (unsigned short)(r >> 16);
}

__device__ __forceinline__ float sigmoidf_(float x) { return 1.0f / (1.0f + expf(-x)); }

// ---------------------------------------------------------------------------
// Wave-level GEMV over 16 output rows using WMMA bf16 16x16x32.
//   out[row0+n] = bias[row0+n] + sum_k W[row0+n][k] * x[k]
// A fragment (16x32, all M rows identical = x):
//   lane L (half = L>>4): elems 0..15 = x[kb + 16*half + 0..15]
// B fragment (32x16): lane L -> column n = L&15:
//   elems 0..15 = W[row0+n][kb + 16*half + 0..15]   (two contiguous b128 loads)
// D: every lane's acc[0] holds out column n = L&15 (all M rows identical).
// ---------------------------------------------------------------------------
__device__ __forceinline__ void wmma_gemv16(const unsigned short* __restrict__ W,
                                            int ldw, int K,
                                            const unsigned short* xlds,
                                            const float* __restrict__ bias,
                                            float* __restrict__ out,
                                            int row0)
{
    const int lane = threadIdx.x & 31;
    const int n    = lane & 15;
    const int half = lane >> 4;
    const unsigned short* wrow = W + (size_t)(row0 + n) * (size_t)ldw;

    v8f acc = {0.f, 0.f, 0.f, 0.f, 0.f, 0.f, 0.f, 0.f};
    for (int kb = 0; kb < K; kb += 32) {
        BFrag a, b;
        a.q[0] = *reinterpret_cast<const u32x4*>(xlds + kb + 16 * half);
        a.q[1] = *reinterpret_cast<const u32x4*>(xlds + kb + 16 * half + 8);
        const unsigned short* wp = wrow + kb + 16 * half;
        b.q[0] = *reinterpret_cast<const u32x4*>(wp);
        b.q[1] = *reinterpret_cast<const u32x4*>(wp + 8);
        acc = __builtin_amdgcn_wmma_f32_16x16x32_bf16(
            /*neg_a=*/false, a.v, /*neg_b=*/false, b.v,
            /*c_mod=*/(short)0, acc, /*reuse_a=*/false, /*reuse_b=*/false);
    }
    if (lane < 16) out[row0 + n] = acc[0] + bias[row0 + n];
}

// --------------------------- one-time prep kernels -------------------------

__global__ void k_f2bf(const float* __restrict__ in, unsigned short* __restrict__ out,
                       long long n)
{
    long long i = (long long)blockIdx.x * blockDim.x + threadIdx.x;
    long long stride = (long long)gridDim.x * blockDim.x;
    for (; i < n; i += stride) out[i] = f2bf(in[i]);
}

// enc_proj[s][j] = dot(W_a[j,:], enc[s,:]) + b_a[j]     (64 x 1024)
__global__ void k_encproj(const float* __restrict__ enc, const float* __restrict__ Wa,
                          const float* __restrict__ ba, float* __restrict__ ep)
{
    int id = blockIdx.x * blockDim.x + threadIdx.x;   // 65536 threads
    int s = id >> 10, j = id & 1023;
    const float* er = enc + s * H_SIZE;
    const float* wr = Wa + j * H_SIZE;
    float acc = 0.f;
    for (int k = 0; k < H_SIZE; ++k) acc += er[k] * wr[k];
    ep[id] = acc + ba[j];
}

__global__ void k_init(const float* __restrict__ h0, float* __restrict__ h,
                       float* __restrict__ ctx, int* __restrict__ tok)
{
    int i = blockIdx.x * blockDim.x + threadIdx.x;
    if (i < H_SIZE) { h[i] = h0[i]; ctx[i] = 0.f; }
    if (i == 0) *tok = SOS_TOK;
}

// ------------------------------ per-step kernels ---------------------------

// Blocks 0..23: gi = W_ih_bf @ bf16([emb[tok]; ctx]) + b_ih   (3072 rows, K=1536)
// Blocks 24..47: gh = W_hh_bf @ bf16(h) + b_hh                (3072 rows, K=1024)
__global__ void k_gru_gemv(const unsigned short* __restrict__ Wih,
                           const unsigned short* __restrict__ Whh,
                           const float* __restrict__ emb, const int* __restrict__ tok,
                           const float* __restrict__ ctx, const float* __restrict__ h,
                           const float* __restrict__ b_ih, const float* __restrict__ b_hh,
                           float* __restrict__ gi, float* __restrict__ gh)
{
    __shared__ alignas(16) unsigned short xl[E_SIZE + H_SIZE];   // 1536 bf16
    const bool isIH = (blockIdx.x < 24);
    if (isIH) {
        int t = *tok;
        const float* erow = emb + (size_t)t * E_SIZE;
        for (int i = threadIdx.x; i < E_SIZE; i += blockDim.x) xl[i] = f2bf(erow[i]);
        for (int i = threadIdx.x; i < H_SIZE; i += blockDim.x) xl[E_SIZE + i] = f2bf(ctx[i]);
    } else {
        for (int i = threadIdx.x; i < H_SIZE; i += blockDim.x) xl[i] = f2bf(h[i]);
    }
    __syncthreads();
    const int wave = threadIdx.x >> 5;                // 8 waves x 16 rows = 128 rows/block
    if (isIH) {
        int row0 = blockIdx.x * 128 + wave * 16;
        wmma_gemv16(Wih, E_SIZE + H_SIZE, E_SIZE + H_SIZE, xl, b_ih, gi, row0);
    } else {
        int row0 = (blockIdx.x - 24) * 128 + wave * 16;
        wmma_gemv16(Whh, H_SIZE, H_SIZE, xl, b_hh, gh, row0);
    }
}

// One block, 1024 threads: GRU gates -> h_new (in place), attention -> ctx (in place).
__global__ void k_gates_attn(const float* __restrict__ gi, const float* __restrict__ gh,
                             float* __restrict__ h, const float* __restrict__ ep,
                             const float* __restrict__ enc, float* __restrict__ ctx)
{
    __shared__ float sh[H_SIZE];      // h_new
    __shared__ float sp[H_SIZE];      // score partials
    __shared__ float ssc[S_SIZE];     // scores
    __shared__ float sattn[S_SIZE];   // softmax weights
    const int i = threadIdx.x;

    float r  = sigmoidf_(gi[i] + gh[i]);
    float z  = sigmoidf_(gi[H_SIZE + i] + gh[H_SIZE + i]);
    float nn = tanhf(gi[2 * H_SIZE + i] + r * gh[2 * H_SIZE + i]);
    float hn = (1.f - z) * nn + z * h[i];
    h[i] = hn;
    sh[i] = hn;
    __syncthreads();

    {   // scores[s] = dot(enc_proj[s,:], h_new); 16 lanes per s
        int s = i >> 4, l = i & 15;
        const float* er = ep + s * H_SIZE;
        float p = 0.f;
        for (int k = l; k < H_SIZE; k += 16) p += er[k] * sh[k];
        sp[i] = p;
    }
    __syncthreads();
    if (i < S_SIZE) {
        float sc = 0.f;
        for (int l = 0; l < 16; ++l) sc += sp[i * 16 + l];
        ssc[i] = sc;
    }
    __syncthreads();
    if (i == 0) {   // softmax over 64 (trivial)
        float m = ssc[0];
        for (int s = 1; s < S_SIZE; ++s) m = fmaxf(m, ssc[s]);
        float sum = 0.f;
        for (int s = 0; s < S_SIZE; ++s) { float e = expf(ssc[s] - m); sattn[s] = e; sum += e; }
        float inv = 1.f / sum;
        for (int s = 0; s < S_SIZE; ++s) sattn[s] *= inv;
    }
    __syncthreads();
    float c = 0.f;
    for (int s = 0; s < S_SIZE; ++s) c += sattn[s] * enc[s * H_SIZE + i];
    ctx[i] = c;
}

// logits = W_o_bf @ bf16([h_new; ctx_new]) + b_o   (32000 rows, K=2048), 250 blocks
__global__ void k_logits(const unsigned short* __restrict__ Wo,
                         const float* __restrict__ h, const float* __restrict__ ctx,
                         const float* __restrict__ b_o, float* __restrict__ logits)
{
    __shared__ alignas(16) unsigned short xl[2 * H_SIZE];
    for (int i = threadIdx.x; i < H_SIZE; i += blockDim.x) {
        xl[i] = f2bf(h[i]);
        xl[H_SIZE + i] = f2bf(ctx[i]);
    }
    __syncthreads();
    int row0 = blockIdx.x * 128 + (threadIdx.x >> 5) * 16;
    wmma_gemv16(Wo, 2 * H_SIZE, 2 * H_SIZE, xl, b_o, logits, row0);
}

// One block: max/argmax, sum(exp), write logp row + word; feed token back.
__global__ void k_logsoftmax(const float* __restrict__ logits,
                             float* __restrict__ out_logp, float* __restrict__ out_word,
                             int* __restrict__ tok)
{
    __shared__ float sred[1024];
    __shared__ int   sidx[1024];
    __shared__ float s_m, s_lz;
    __shared__ int   s_bi;
    const int tid = threadIdx.x;

    float lm = -3.402823466e38f; int li = 0;
    for (int i = tid; i < V_SIZE; i += 1024) {
        float v = logits[i];
        if (v > lm) { lm = v; li = i; }
    }
    sred[tid] = lm; sidx[tid] = li;
    __syncthreads();
    for (int s = 512; s > 0; s >>= 1) {
        if (tid < s) {
            float v2 = sred[tid + s]; int i2 = sidx[tid + s];
            if (v2 > sred[tid] || (v2 == sred[tid] && i2 < sidx[tid])) {
                sred[tid] = v2; sidx[tid] = i2;
            }
        }
        __syncthreads();
    }
    if (tid == 0) { s_m = sred[0]; s_bi = sidx[0]; }
    __syncthreads();

    const float m = s_m;
    float ls = 0.f;
    for (int i = tid; i < V_SIZE; i += 1024) ls += expf(logits[i] - m);
    sred[tid] = ls;
    __syncthreads();
    for (int s = 512; s > 0; s >>= 1) {
        if (tid < s) sred[tid] += sred[tid + s];
        __syncthreads();
    }
    if (tid == 0) s_lz = m + logf(sred[0]);
    __syncthreads();

    const float lz = s_lz;
    for (int i = tid; i < V_SIZE; i += 1024) out_logp[i] = logits[i] - lz;
    if (tid == 0) { *tok = s_bi; out_word[0] = (float)s_bi; }
}

// ------------------------------- host launcher -----------------------------

extern "C" void kernel_launch(void* const* d_in, const int* in_sizes, int n_in,
                              void* d_out, int out_size, void* d_ws, size_t ws_size,
                              hipStream_t stream)
{
    (void)in_sizes; (void)n_in; (void)out_size; (void)ws_size;

    const float* init_hidden = (const float*)d_in[0];
    const float* enc         = (const float*)d_in[1];
    // d_in[2] = tgt_len (int scalar) == 32, fixed by setup
    const float* emb         = (const float*)d_in[3];
    const float* W_a         = (const float*)d_in[4];
    const float* b_a         = (const float*)d_in[5];
    const float* W_ih        = (const float*)d_in[6];
    const float* W_hh        = (const float*)d_in[7];
    const float* b_ih        = (const float*)d_in[8];
    const float* b_hh        = (const float*)d_in[9];
    const float* W_o         = (const float*)d_in[10];
    const float* b_o         = (const float*)d_in[11];
    float* out = (float*)d_out;

    char* ws = (char*)d_ws;
    size_t cur = 0;
    auto walloc = [&](size_t bytes) -> char* {
        char* p = ws + cur;
        cur = (cur + bytes + 255) & ~(size_t)255;
        return p;
    };

    unsigned short* Wo_bf  = (unsigned short*)walloc((size_t)V_SIZE * 2 * H_SIZE * 2);
    unsigned short* Wih_bf = (unsigned short*)walloc((size_t)3 * H_SIZE * (E_SIZE + H_SIZE) * 2);
    unsigned short* Whh_bf = (unsigned short*)walloc((size_t)3 * H_SIZE * H_SIZE * 2);
    float* ep     = (float*)walloc((size_t)S_SIZE * H_SIZE * 4);
    float* h      = (float*)walloc(H_SIZE * 4);
    float* ctx    = (float*)walloc(H_SIZE * 4);
    float* gi     = (float*)walloc(3 * H_SIZE * 4);
    float* gh     = (float*)walloc(3 * H_SIZE * 4);
    float* logits = (float*)walloc((size_t)V_SIZE * 4);
    int*   tok    = (int*)walloc(256);

    // One-time: bf16 weight images (W_o_bf = 131MB -> L2 resident), enc_proj, state.
    k_f2bf<<<4096, 256, 0, stream>>>(W_o,  Wo_bf,  (long long)V_SIZE * 2 * H_SIZE);
    k_f2bf<<<1024, 256, 0, stream>>>(W_ih, Wih_bf, (long long)3 * H_SIZE * (E_SIZE + H_SIZE));
    k_f2bf<<<1024, 256, 0, stream>>>(W_hh, Whh_bf, (long long)3 * H_SIZE * H_SIZE);
    k_encproj<<<(S_SIZE * H_SIZE) / 256, 256, 0, stream>>>(enc, W_a, b_a, ep);
    k_init<<<4, 256, 0, stream>>>(init_hidden, h, ctx, tok);

    for (int t = 0; t < T_STEPS; ++t) {
        k_gru_gemv<<<48, 256, 0, stream>>>(Wih_bf, Whh_bf, emb, tok, ctx, h,
                                           b_ih, b_hh, gi, gh);
        k_gates_attn<<<1, 1024, 0, stream>>>(gi, gh, h, ep, enc, ctx);
        k_logits<<<250, 256, 0, stream>>>(Wo_bf, h, ctx, b_o, logits);
        k_logsoftmax<<<1, 1024, 0, stream>>>(logits,
                                             out + (size_t)t * V_SIZE,
                                             out + (size_t)T_STEPS * V_SIZE + t,
                                             tok);
    }
}
